// MultiHeadAttention_32066225832689
// MI455X (gfx1250) — matrix operand order repaired and checked by reading, not compiled, not run on hardware
//
#include <hip/hip_runtime.h>
#include <hip/hip_bf16.h>

typedef __attribute__((ext_vector_type(16))) __bf16 v16bf;
typedef __attribute__((ext_vector_type(8)))  __bf16 v8bf;
typedef __attribute__((ext_vector_type(8)))  float  v8f;

#define D_DIM 1024
#define S_DIM 2048
#define B_DIM 2
#define NROWS (B_DIM * S_DIM)   // 4096

// ---------------- WMMA fragment loaders (wave32, CDNA5 layouts) ----------------

// A-matrix 16x32 bf16, source row-major with leading dim `ld`, p -> element (0, 0) of tile.
// Lane L: row = L&15, half = L>>4; elems 0..7 = K[half*8 .. +8), elems 8..15 = K[16+half*8 .. +8).
__device__ __forceinline__ v16bf load_a_bf16(const __bf16* p, int ld) {
  const int lane = threadIdx.x & 31;
  const int half = lane >> 4, row = lane & 15;
  const __bf16* r = p + (size_t)row * ld;
  v8bf c0 = *reinterpret_cast<const v8bf*>(r + half * 8);
  v8bf c1 = *reinterpret_cast<const v8bf*>(r + 16 + half * 8);
  return __builtin_shufflevector(c0, c1, 0,1,2,3,4,5,6,7,8,9,10,11,12,13,14,15);
}

// Same A layout but source is fp32 (e.g. LDS softmax buffer); convert to bf16 on the fly.
__device__ __forceinline__ v16bf load_a_f32(const float* p, int ld) {
  const int lane = threadIdx.x & 31;
  const int half = lane >> 4, row = lane & 15;
  const float* r = p + (size_t)row * ld;
  v8f c0 = *reinterpret_cast<const v8f*>(r + half * 8);
  v8f c1 = *reinterpret_cast<const v8f*>(r + 16 + half * 8);
  v16bf o;
#pragma unroll
  for (int i = 0; i < 8; ++i) { o[i] = (__bf16)c0[i]; o[i + 8] = (__bf16)c1[i]; }
  return o;
}

// B-matrix 32x16 bf16 from transposed ([N][K] row-major) storage, p -> element (n0, k0).
// Lane L: col = L&15, half = L>>4; elems 0..7 = K[half*8 .. +8), elems 8..15 = K[16+half*8 .. +8).
__device__ __forceinline__ v16bf load_bt(const __bf16* p, int ld) {
  const int lane = threadIdx.x & 31;
  const int half = lane >> 4, col = lane & 15;
  const __bf16* r = p + (size_t)col * ld;
  v8bf c0 = *reinterpret_cast<const v8bf*>(r + half * 8);
  v8bf c1 = *reinterpret_cast<const v8bf*>(r + 16 + half * 8);
  return __builtin_shufflevector(c0, c1, 0,1,2,3,4,5,6,7,8,9,10,11,12,13,14,15);
}

// ---------------- conversion kernels ----------------

__global__ void cvt_f32_to_bf16(const float* __restrict__ src, __bf16* __restrict__ dst, int n) {
  int i = blockIdx.x * blockDim.x + threadIdx.x;
  if (i < n) dst[i] = (__bf16)src[i];
}

// WT[j][i] = bf16(W[i][j]), D_DIM x D_DIM, 32x32 LDS tiles.
__global__ __launch_bounds__(256) void transpose_to_bf16(const float* __restrict__ W,
                                                         __bf16* __restrict__ WT) {
  __shared__ __bf16 tile[32][33];
  const int bx = blockIdx.x * 32, by = blockIdx.y * 32;
  const int tx = threadIdx.x & 31, ty = threadIdx.x >> 5;   // ty = 0..7
#pragma unroll
  for (int r = ty; r < 32; r += 8)
    tile[r][tx] = (__bf16)W[(size_t)(by + r) * D_DIM + (bx + tx)];
  __syncthreads();
#pragma unroll
  for (int r = ty; r < 32; r += 8)
    WT[(size_t)(bx + r) * D_DIM + (by + tx)] = tile[tx][r];
}

// ---------------- fused QKV projection: relu(x@W + b) ----------------
// grid = (NROWS/16, D_DIM/512, 3); block = 256 = 8 waves.
// A strip (16 x 1024 bf16, 32 KB) staged once in LDS and shared by all waves;
// each wave register-blocks 4 adjacent 16x16 output tiles (one A frag -> 4 WMMAs).

__global__ __launch_bounds__(256) void qkv_proj(
    const __bf16* __restrict__ xb,
    const __bf16* __restrict__ WqT, const __bf16* __restrict__ WkT, const __bf16* __restrict__ WvT,
    const float* __restrict__ bq, const float* __restrict__ bk, const float* __restrict__ bv,
    __bf16* __restrict__ qb, __bf16* __restrict__ kb, __bf16* __restrict__ vT) {
  __shared__ __bf16 As[16 * D_DIM];

  const int which = blockIdx.z;
  const __bf16* WT   = (which == 0) ? WqT : (which == 1) ? WkT : WvT;
  const float*  bias = (which == 0) ? bq : (which == 1) ? bk : bv;
  const int wave = threadIdx.x >> 5;
  const int m0 = blockIdx.x * 16;
  const int n0w = blockIdx.y * 512 + wave * 64;   // 4 tiles of 16 columns per wave

  // cooperative stage of the contiguous 32 KB A strip
  {
    const v8bf* src = reinterpret_cast<const v8bf*>(xb + (size_t)m0 * D_DIM);
    v8bf* dst = reinterpret_cast<v8bf*>(As);
#pragma unroll
    for (int i = threadIdx.x; i < (16 * D_DIM) / 8; i += 256) dst[i] = src[i];
  }
  __syncthreads();

  v8f acc[4] = {};
  for (int k0 = 0; k0 < D_DIM; k0 += 32) {
    v16bf a = load_a_bf16(As + k0, D_DIM);        // LDS
#pragma unroll
    for (int t = 0; t < 4; ++t) {
      v16bf b = load_bt(WT + (size_t)(n0w + t * 16) * D_DIM + k0, D_DIM);
      acc[t] = __builtin_amdgcn_wmma_f32_16x16x32_bf16(false, a, false, b, (short)0, acc[t],
                                                       false, false);
    }
  }

  const int lane = threadIdx.x & 31;
  const int col = lane & 15, rowbase = (lane >> 4) * 8;
#pragma unroll
  for (int t = 0; t < 4; ++t) {
    const int n0 = n0w + t * 16;
    const float bvv = bias[n0 + col];
#pragma unroll
    for (int j = 0; j < 8; ++j) {
      float r = fmaxf(acc[t][j] + bvv, 0.0f);
      const int g = m0 + rowbase + j;
      if (which == 2) {  // v stored transposed: vT[b][d][s]
        const int bb = g / S_DIM, s = g % S_DIM;
        vT[(size_t)bb * D_DIM * S_DIM + (size_t)(n0 + col) * S_DIM + s] = (__bf16)r;
      } else {
        __bf16* o = (which == 0) ? qb : kb;
        o[(size_t)g * D_DIM + (n0 + col)] = (__bf16)r;
      }
    }
  }
}

// ---------------- attention: scores -> softmax -> PV ----------------
// One 16-row query tile per block; 256 threads = 8 waves.
// LDS: 16 x 2048 fp32 score/prob buffer (128 KB) within the 320 KB WGP budget.

__global__ __launch_bounds__(256) void attn_kernel(
    const __bf16* __restrict__ qb, const __bf16* __restrict__ kb,
    const __bf16* __restrict__ vT, __bf16* __restrict__ yb) {
  __shared__ float Sbuf[16 * S_DIM];
  __shared__ float red[16 * 16];
  __shared__ float rowmax[16];
  __shared__ float rowsum[16];

  const int qt = blockIdx.x;                 // 0..255
  const int bb = qt / (S_DIM / 16);
  const int q0 = (qt % (S_DIM / 16)) * 16;   // within-batch query base
  const int gq = bb * S_DIM + q0;            // global query row base
  const int wave = threadIdx.x >> 5;
  const int lane = threadIdx.x & 31;
  const int col = lane & 15, rowbase = (lane >> 4) * 8;

  // ---- Phase 1: raw scaled + causal-masked scores into LDS ----
  // Each wave owns 256 keys = 4 groups of 4 key tiles; one A frag feeds 4 WMMAs.
  for (int g = 0; g < 4; ++g) {
    const int gbase = wave * 256 + g * 64;   // key-group base (within batch)
    if (gbase > q0 + 15) {                   // whole group causally masked: skip WMMA
#pragma unroll
      for (int t = 0; t < 4; ++t)
#pragma unroll
        for (int j = 0; j < 8; ++j)
          Sbuf[(rowbase + j) * S_DIM + gbase + t * 16 + col] = -1.0e30f;
      continue;
    }
    v8f acc[4] = {};
    for (int k0 = 0; k0 < D_DIM; k0 += 32) {
      v16bf a = load_a_bf16(qb + (size_t)gq * D_DIM + k0, D_DIM);
#pragma unroll
      for (int t = 0; t < 4; ++t) {
        v16bf b = load_bt(kb + (size_t)(bb * S_DIM + gbase + t * 16) * D_DIM + k0, D_DIM);
        acc[t] = __builtin_amdgcn_wmma_f32_16x16x32_bf16(false, a, false, b, (short)0, acc[t],
                                                         false, false);
      }
    }
#pragma unroll
    for (int t = 0; t < 4; ++t) {
      const int key = gbase + t * 16 + col;
#pragma unroll
      for (int j = 0; j < 8; ++j) {
        const int qr = q0 + rowbase + j;
        float s = acc[t][j] * 0.03125f;      // 1/sqrt(1024)
        if (key > qr) s -= 10000.0f;         // reference causal penalty
        Sbuf[(rowbase + j) * S_DIM + key] = s;
      }
    }
  }
  __syncthreads();

  // ---- Phase 2: row-wise stable softmax (unnormalized exp; keep sums) ----
  const int r = threadIdx.x >> 4;            // row 0..15
  const int t16 = threadIdx.x & 15;
  float m = -3.0e38f;
  for (int c = t16; c < S_DIM; c += 16) m = fmaxf(m, Sbuf[r * S_DIM + c]);
  red[r * 16 + t16] = m;
  __syncthreads();
  if (t16 == 0) {
    float mm = -3.0e38f;
#pragma unroll
    for (int i = 0; i < 16; ++i) mm = fmaxf(mm, red[r * 16 + i]);
    rowmax[r] = mm;
  }
  __syncthreads();
  const float mm = rowmax[r];
  float ssum = 0.0f;
  for (int c = t16; c < S_DIM; c += 16) {
    float e = __expf(Sbuf[r * S_DIM + c] - mm);
    Sbuf[r * S_DIM + c] = e;
    ssum += e;
  }
  red[r * 16 + t16] = ssum;
  __syncthreads();
  if (t16 == 0) {
    float sm = 0.0f;
#pragma unroll
    for (int i = 0; i < 16; ++i) sm += red[r * 16 + i];
    rowsum[r] = sm;
  }
  __syncthreads();

  // ---- Phase 3: y = P @ V; one LDS A frag feeds 8 WMMAs; fold 1/rowsum in ----
  {
    v8f acc[8] = {};
    const size_t vbase = (size_t)bb * D_DIM * S_DIM;
    for (int k0 = 0; k0 < S_DIM; k0 += 32) {
      v16bf a = load_a_f32(Sbuf + k0, S_DIM);                // LDS -> bf16 frag (once)
#pragma unroll
      for (int ct = 0; ct < 8; ++ct) {
        const int n0 = wave * 128 + ct * 16;                 // output feature-column base
        v16bf b = load_bt(vT + vbase + (size_t)n0 * S_DIM + k0, S_DIM);
        acc[ct] = __builtin_amdgcn_wmma_f32_16x16x32_bf16(false, a, false, b, (short)0, acc[ct],
                                                          false, false);
      }
    }
#pragma unroll
    for (int ct = 0; ct < 8; ++ct) {
      const int n0 = wave * 128 + ct * 16;
#pragma unroll
      for (int j = 0; j < 8; ++j) {
        const float y = acc[ct][j] / rowsum[rowbase + j];
        yb[(size_t)(gq + rowbase + j) * D_DIM + (n0 + col)] = (__bf16)y;
      }
    }
  }
}

// ---------------- output projection: relu(y@Wo + bo) -> fp32 ----------------
// Same LDS-staged, 4-tile register-blocked structure as qkv_proj.

__global__ __launch_bounds__(256) void out_proj(
    const __bf16* __restrict__ yb, const __bf16* __restrict__ WoT,
    const float* __restrict__ bo, float* __restrict__ out) {
  __shared__ __bf16 As[16 * D_DIM];

  const int wave = threadIdx.x >> 5;
  const int m0 = blockIdx.x * 16;
  const int n0w = blockIdx.y * 512 + wave * 64;

  {
    const v8bf* src = reinterpret_cast<const v8bf*>(yb + (size_t)m0 * D_DIM);
    v8bf* dst = reinterpret_cast<v8bf*>(As);
#pragma unroll
    for (int i = threadIdx.x; i < (16 * D_DIM) / 8; i += 256) dst[i] = src[i];
  }
  __syncthreads();

  v8f acc[4] = {};
  for (int k0 = 0; k0 < D_DIM; k0 += 32) {
    v16bf a = load_a_bf16(As + k0, D_DIM);
#pragma unroll
    for (int t = 0; t < 4; ++t) {
      v16bf b = load_bt(WoT + (size_t)(n0w + t * 16) * D_DIM + k0, D_DIM);
      acc[t] = __builtin_amdgcn_wmma_f32_16x16x32_bf16(false, a, false, b, (short)0, acc[t],
                                                       false, false);
    }
  }

  const int lane = threadIdx.x & 31;
  const int col = lane & 15, rowbase = (lane >> 4) * 8;
#pragma unroll
  for (int t = 0; t < 4; ++t) {
    const int n0 = n0w + t * 16;
    const float bvv = bo[n0 + col];
#pragma unroll
    for (int j = 0; j < 8; ++j)
      out[(size_t)(m0 + rowbase + j) * D_DIM + (n0 + col)] = fmaxf(acc[t][j] + bvv, 0.0f);
  }
}

// ---------------- host launcher ----------------

extern "C" void kernel_launch(void* const* d_in, const int* in_sizes, int n_in,
                              void* d_out, int out_size, void* d_ws, size_t ws_size,
                              hipStream_t stream) {
  const float* x  = (const float*)d_in[0];
  const float* Wq = (const float*)d_in[1];
  const float* bq = (const float*)d_in[2];
  const float* Wk = (const float*)d_in[3];
  const float* bk = (const float*)d_in[4];
  const float* Wv = (const float*)d_in[5];
  const float* bv = (const float*)d_in[6];
  const float* Wo = (const float*)d_in[7];
  const float* bo = (const float*)d_in[8];
  float* out = (float*)d_out;

  char* ws = (char*)d_ws;
  const size_t MB = 1024u * 1024u;
  __bf16* xb  = (__bf16*)(ws);                 // 8 MB   (4096x1024 bf16)
  __bf16* WqT = (__bf16*)(ws + 8  * MB);       // 2 MB each
  __bf16* WkT = (__bf16*)(ws + 10 * MB);
  __bf16* WvT = (__bf16*)(ws + 12 * MB);
  __bf16* WoT = (__bf16*)(ws + 14 * MB);
  __bf16* qb  = (__bf16*)(ws + 16 * MB);       // 8 MB
  __bf16* kb  = (__bf16*)(ws + 24 * MB);       // 8 MB
  __bf16* vT  = (__bf16*)(ws + 32 * MB);       // 8 MB (per-batch transposed)
  __bf16* yb  = (__bf16*)(ws + 40 * MB);       // 8 MB

  const int n_x = NROWS * D_DIM;
  cvt_f32_to_bf16<<<n_x / 256, 256, 0, stream>>>(x, xb, n_x);

  dim3 tg(D_DIM / 32, D_DIM / 32);
  transpose_to_bf16<<<tg, 256, 0, stream>>>(Wq, WqT);
  transpose_to_bf16<<<tg, 256, 0, stream>>>(Wk, WkT);
  transpose_to_bf16<<<tg, 256, 0, stream>>>(Wv, WvT);
  transpose_to_bf16<<<tg, 256, 0, stream>>>(Wo, WoT);

  qkv_proj<<<dim3(NROWS / 16, D_DIM / 512, 3), 256, 0, stream>>>(
      xb, WqT, WkT, WvT, bq, bk, bv, qb, kb, vT);

  attn_kernel<<<dim3(NROWS / 16), 256, 0, stream>>>(qb, kb, vT, yb);

  out_proj<<<dim3(NROWS / 16, D_DIM / 512), 256, 0, stream>>>(yb, WoT, bo, out);
}
